// RNNNetwork_48782238547986
// MI455X (gfx1250) — matrix operand order repaired and checked by reading, not compiled, not run on hardware
//
#include <hip/hip_runtime.h>

#define BATCH 8192
#define TSTEPS 512
#define HDIM 32
#define ODIM 4

typedef __attribute__((ext_vector_type(16))) _Float16 v16h;
typedef __attribute__((ext_vector_type(8)))  float    v8f;
typedef __attribute__((ext_vector_type(4)))  float    v4f;
typedef __attribute__((ext_vector_type(2)))  float    v2f;
typedef __attribute__((ext_vector_type(8)))  unsigned v8u;
typedef __attribute__((ext_vector_type(2)))  unsigned v2u;

#if __has_builtin(__builtin_amdgcn_wmma_f32_16x16x4_f32)
#define HAVE_K4 1
#else
#define HAVE_K4 0
#endif

#if __has_builtin(__builtin_amdgcn_permlane16_swap)
#define HAVE_PLSWAP 1
#else
#define HAVE_PLSWAP 0
#endif

static __device__ inline float fast_tanh(float x) {
#if __has_builtin(__builtin_amdgcn_tanhf)
  return __builtin_amdgcn_tanhf(x);   // v_tanh_f32 (gfx1250 trans op)
#else
  return tanhf(x);
#endif
}

static __device__ inline v8f tanh8(v8f x) {
  v8f r;
#pragma unroll
  for (int i = 0; i < 8; ++i) r[i] = fast_tanh(x[i]);
  return r;
}

static __device__ inline unsigned pk(float a, float b) {
  return __builtin_bit_cast(unsigned, __builtin_amdgcn_cvt_pkrtz(a, b));
}

// Build a 16x32 f16 A-fragment from one weight row (row-major, 32 cols).
// A layout: lane<16: M=lane, K in {0..15}; lane>=16: M=lane-16, K in {16..31}.
static __device__ inline v16h load_A16(const float* row, bool valid, int lane) {
  const int ko = (lane < 16) ? 0 : 16;
  v8u r;
#pragma unroll
  for (int v = 0; v < 4; ++v) {
    float a0 = valid ? row[ko + 2 * v]      : 0.f;
    float a1 = valid ? row[ko + 2 * v + 1]  : 0.f;
    float c0 = valid ? row[8 + ko + 2 * v]     : 0.f;
    float c1 = valid ? row[8 + ko + 2 * v + 1] : 0.f;
    r[v]     = pk(a0, a1);
    r[v + 4] = pk(c0, c1);
  }
  return __builtin_bit_cast(v16h, r);
}

// Bias as a C/D fragment: VGPR r holds M = r (+8 for upper lane half).
static __device__ inline v8f bias_frag(const float* a, const float* b, int lane) {
  v8f f;
  const int off = (lane < 16) ? 0 : 8;
#pragma unroll
  for (int r = 0; r < 8; ++r) f[r] = a[off + r] + b[off + r];
  return f;
}

// Convert two tanh'ed D accumulators (rows 0-15 in d0, rows 16-31 in d1) into the
// 32x16 f16 B fragment for the next step's WMMA.
// Wanted: B[j] = {q0.lo16 , q1.lo16}, B[4+j] = {q0.hi16 , q1.hi16} (per lane halves).
// v_permlane16_swap_b32(q0, q1) produces exactly that pair in one VALU op; fallback
// uses ds_swizzle SWAPX16 + cndmask.
static __device__ inline v16h pack_B(v8f d0, v8f d1, bool lo) {
  unsigned q0[4], q1[4];
#pragma unroll
  for (int j = 0; j < 4; ++j) {
    q0[j] = pk(d0[2 * j], d0[2 * j + 1]);
    q1[j] = pk(d1[2 * j], d1[2 * j + 1]);
  }
  v8u b;
#if HAVE_PLSWAP
#pragma unroll
  for (int j = 0; j < 4; ++j) {
    v2u r = __builtin_amdgcn_permlane16_swap(q0[j], q1[j], false, false);
    b[j]     = r[0];  // lanes<16: K=2j,2j+1   | lanes>=16: K=16+2j,17+2j
    b[j + 4] = r[1];  // lanes<16: K=8+2j,9+2j | lanes>=16: K=24+2j,25+2j
  }
#else
#pragma unroll
  for (int j = 0; j < 4; ++j) {
    unsigned s0 = (unsigned)__builtin_amdgcn_ds_swizzle((int)q0[j], 0x401f);
    unsigned s1 = (unsigned)__builtin_amdgcn_ds_swizzle((int)q1[j], 0x401f);
    b[j]     = lo ? q0[j] : s1;
    b[j + 4] = lo ? s0    : q1[j];
  }
#endif
  return __builtin_bit_cast(v16h, b);
}

#define WMMA16(A, Bm, C) \
  __builtin_amdgcn_wmma_f32_16x16x32_f16(false, (A), false, (Bm), (short)0, (C), false, false)

__global__ __launch_bounds__(256, 1) void rnn2_wmma_kernel(
    const float* __restrict__ X,
    const float* __restrict__ Wih0, const float* __restrict__ Whh0,
    const float* __restrict__ bih0, const float* __restrict__ bhh0,
    const float* __restrict__ Wih1, const float* __restrict__ Whh1,
    const float* __restrict__ bih1, const float* __restrict__ bhh1,
    const float* __restrict__ Wll,  const float* __restrict__ bll,
    float* __restrict__ Out) {
  const int lane  = threadIdx.x & 31;
  const int wv    = threadIdx.x >> 5;
  const int tile  = blockIdx.x * 8 + wv;   // one wave = one 16-batch tile
  const int bbase = tile * 16;
  const bool lo   = lane < 16;
  const int  col  = lane & 15;

  // ---- loop-invariant register-resident fragments ----
  v16h A_hh0_lo = load_A16(Whh0 + (size_t)(0  + col) * HDIM, true, lane);
  v16h A_hh0_hi = load_A16(Whh0 + (size_t)(16 + col) * HDIM, true, lane);
  v16h A_ih1_lo = load_A16(Wih1 + (size_t)(0  + col) * HDIM, true, lane);
  v16h A_ih1_hi = load_A16(Wih1 + (size_t)(16 + col) * HDIM, true, lane);
  v16h A_hh1_lo = load_A16(Whh1 + (size_t)(0  + col) * HDIM, true, lane);
  v16h A_hh1_hi = load_A16(Whh1 + (size_t)(16 + col) * HDIM, true, lane);
  v16h A_ll     = load_A16(Wll  + (size_t)col * HDIM, col < ODIM, lane); // rows 4..15 = 0

  v8f Bias0_lo = bias_frag(bih0,      bhh0,      lane);
  v8f Bias0_hi = bias_frag(bih0 + 16, bhh0 + 16, lane);
  v8f Bias1_lo = bias_frag(bih1,      bhh1,      lane);
  v8f Bias1_hi = bias_frag(bih1 + 16, bhh1 + 16, lane);
  v8f BiasLL;
  {
    const int off = lo ? 0 : 8;
#pragma unroll
    for (int r = 0; r < 8; ++r) BiasLL[r] = (off + r < ODIM) ? bll[off + r] : 0.f;
  }

#if HAVE_K4
  // W_ih0 as two 16x4 f32 A fragments (K=4): V0/V1 hold K={0,1} (lo half) or {2,3}.
  const int k0 = lo ? 0 : 2;
  v2f A_ih0_lo, A_ih0_hi;
  A_ih0_lo.x = Wih0[(0  + col) * 4 + k0];
  A_ih0_lo.y = Wih0[(0  + col) * 4 + k0 + 1];
  A_ih0_hi.x = Wih0[(16 + col) * 4 + k0];
  A_ih0_hi.y = Wih0[(16 + col) * 4 + k0 + 1];
  const size_t lanebase = (size_t)(bbase + col) * TSTEPS * 4 + (lo ? 0 : 2);
#else
  float W0a[8][4], W0b[8][4];
  {
    const int off = lo ? 0 : 8;
#pragma unroll
    for (int r = 0; r < 8; ++r)
#pragma unroll
      for (int i = 0; i < 4; ++i) {
        W0a[r][i] = Wih0[(off + r) * 4 + i];
        W0b[r][i] = Wih0[(16 + off + r) * 4 + i];
      }
  }
  const size_t lanebase = (size_t)(bbase + col) * TSTEPS * 4;
#endif

  v16h h0B = __builtin_bit_cast(v16h, (v8u)0u);
  v16h h1B = __builtin_bit_cast(v16h, (v8u)0u);

  float* optr = Out + (size_t)(bbase + col) * (TSTEPS * ODIM);

#if HAVE_K4
  // software-pipelined x operand: xb holds step t, loaded one iteration ahead
  v2f xb;
  {
    const float* p = X + lanebase;
    xb.x = p[0];
    xb.y = p[1];
  }
#else
  v4f xq = *reinterpret_cast<const v4f*>(X + lanebase);
#endif

#pragma unroll 1
  for (int t = 0; t < TSTEPS; ++t) {
    // ---- issue next step's x load early (wrap index keeps it in-bounds) ----
    const int tn = (t + 1) & (TSTEPS - 1);
#if HAVE_K4
    const float* pn = X + lanebase + (size_t)tn * 4;
    v2f xn;
    xn.x = pn[0];
    xn.y = pn[1];
#else
    v4f xqn = *reinterpret_cast<const v4f*>(X + lanebase + (size_t)tn * 4);
#endif
    // deep prefetch: 4 cache lines ahead of use
    __builtin_prefetch(X + lanebase + (size_t)((t + 16) & (TSTEPS - 1)) * 4, 0, 0);

    // ---- layer 0 pre-activation: bias + W_ih0 * x_t  (+ W_hh0 * h0) ----
#if HAVE_K4
    v8f u0a = __builtin_amdgcn_wmma_f32_16x16x4_f32(false, A_ih0_lo, false, xb,
                                                    (short)0, Bias0_lo, false, false);
    v8f u0b = __builtin_amdgcn_wmma_f32_16x16x4_f32(false, A_ih0_hi, false, xb,
                                                    (short)0, Bias0_hi, false, false);
    xb = xn;
#else
    v8f u0a = Bias0_lo, u0b = Bias0_hi;
#pragma unroll
    for (int r = 0; r < 8; ++r)
#pragma unroll
      for (int i = 0; i < 4; ++i) {
        u0a[r] += W0a[r][i] * xq[i];
        u0b[r] += W0b[r][i] * xq[i];
      }
    xq = xqn;
#endif

    u0a = WMMA16(A_hh0_lo, h0B, u0a);
    u0b = WMMA16(A_hh0_hi, h0B, u0b);
    u0a = tanh8(u0a);
    u0b = tanh8(u0b);
    h0B = pack_B(u0a, u0b, lo);

    // ---- layer 1: bias1 + W_hh1*h1_old + W_ih1*h0_new ----
    v8f u1a = WMMA16(A_hh1_lo, h1B, Bias1_lo);
    v8f u1b = WMMA16(A_hh1_hi, h1B, Bias1_hi);
    u1a = WMMA16(A_ih1_lo, h0B, u1a);
    u1b = WMMA16(A_ih1_hi, h0B, u1b);
    u1a = tanh8(u1a);
    u1b = tanh8(u1b);
    h1B = pack_B(u1a, u1b, lo);

    // ---- linear head: rows 0..3 of (W_ll_pad x h1) ----
    v8f oo = WMMA16(A_ll, h1B, BiasLL);
    if (lo) {
      v4f ov;
      ov.x = oo[0];
      ov.y = oo[1];
      ov.z = oo[2];
      ov.w = oo[3];
      *reinterpret_cast<v4f*>(optr) = ov;   // out[b][t*4 .. t*4+3], b128 store
    }
    optr += 4;
  }
}

extern "C" void kernel_launch(void* const* d_in, const int* in_sizes, int n_in,
                              void* d_out, int out_size, void* d_ws, size_t ws_size,
                              hipStream_t stream) {
  const float* X    = (const float*)d_in[0];
  const float* Wih0 = (const float*)d_in[1];
  const float* Whh0 = (const float*)d_in[2];
  const float* bih0 = (const float*)d_in[3];
  const float* bhh0 = (const float*)d_in[4];
  const float* Wih1 = (const float*)d_in[5];
  const float* Whh1 = (const float*)d_in[6];
  const float* bih1 = (const float*)d_in[7];
  const float* bhh1 = (const float*)d_in[8];
  const float* Wll  = (const float*)d_in[9];
  const float* bll  = (const float*)d_in[10];
  float* Out = (float*)d_out;

  const int tiles  = BATCH / 16;         // 512 independent waves
  const int blocks = tiles / 8;          // 8 waves (256 threads) per block
  rnn2_wmma_kernel<<<blocks, 256, 0, stream>>>(X, Wih0, Whh0, bih0, bhh0,
                                               Wih1, Whh1, bih1, bhh1,
                                               Wll, bll, Out);
}
